// WavenetGenerator_80822694576134
// MI455X (gfx1250) — matrix operand order repaired and checked by reading, not compiled, not run on hardware
//
#include <hip/hip_runtime.h>
#include <math.h>

// ---------------------------------------------------------------------------
// WaveNet autoregressive generator for MI455X (gfx1250, wave32, WMMA).
// One persistent 16-wave workgroup; weights pre-packed once into CDNA5 WMMA
// B-fragment layout (f16) in workspace; activations staged in LDS with M
// padded 8->16; all GEMMs via v_wmma_f32_16x16x32_f16; global_prefetch_b8
// used to hide L2 latency of ring lines / first weight fragments.
// ---------------------------------------------------------------------------

typedef __attribute__((ext_vector_type(16))) _Float16 v16h;
typedef __attribute__((ext_vector_type(8)))  float    v8f;

#define WN_B   8      // batch
#define WN_R   128    // residual channels
#define WN_S   128    // skip channels
#define WN_Q   256    // logits
#define WN_C   64     // cond channels
#define WN_L   20     // layers
#define WN_T   512    // timesteps (TE*CR = 8*64)
#define RST    132    // LDS stride of 128-wide buffers (== 4 mod 64)
#define ZST    260    // LDS stride of z/logits buffer  (== 4 mod 64)
#define CST    68     // LDS stride of cond buffer      (== 4 mod 64)

#define RING_FLOATS (2046 * WN_B * WN_R)   // sum(dil)=2046 -> 2,095,104 f32
#define RING_BYTES  ((size_t)RING_FLOATS * 4)

// packed-weight fragment regions (each fragment = 32 lanes * 16 f16 = 1 KB)
#define FR_RS   3200      // 20*16*10 z-fragments first
#define FR_SKC  4480      // + 20*16*4 res/skip
#define FR_FC   4512      // + 8*4 skip-conv-in
#define FR_LOG  4560      // + 8*6 fc(+cond)
#define NFRAG   4624      // + 16*4 logits

__device__ const int kDIL[WN_L]  = {1,2,4,8,16,32,64,128,256,512,
                                    1,2,4,8,16,32,64,128,256,512};
// ring offsets in units of B*R = 1024 floats (prefix sums of kDIL)
__device__ const int kROFF[WN_L] = {0,1,3,7,15,31,63,127,255,511,
                                    1023,1024,1026,1030,1038,1054,1086,1150,1278,1534};

// --------------------------- WMMA helpers ----------------------------------

__device__ inline v8f wn_wmma(v16h a, v16h b, v8f c) {
  // D = A(16x32 f16) * B(32x16 f16) + C(16x16 f32)
  return __builtin_amdgcn_wmma_f32_16x16x32_f16(
      /*neg_a=*/false, a, /*neg_b=*/false, b,
      /*c_mod=*/(short)0, c, /*reuse_a=*/false, /*reuse_b=*/false);
}

// Build A fragment (16x32, f16) from an fp32 LDS tile; rows 8..15 must be 0
// in LDS (batch padding).  Layout per ISA 7.12.2 (16-bit A, 16x32):
//   lane<16 : VGPR j<4 -> K = k0+2j,2j+1 ; j>=4 -> K = k0+16+2(j-4)
//   lane>=16: same with K offset +8
__device__ inline v16h wn_build_a(const float* __restrict__ src, int stride,
                                  int k0, int lane) {
  const int row = lane & 15;
  const int kb  = k0 + ((lane >> 4) << 3);
  const float* p = src + row * stride + kb;
  v16h a;
#pragma unroll
  for (int j = 0; j < 8; ++j) {
    const int kk = (j < 4) ? (2 * j) : (16 + 2 * (j - 4));
    a[2 * j]     = (_Float16)p[kk];
    a[2 * j + 1] = (_Float16)p[kk + 1];
  }
  return a;
}

// B fragment pre-packed in workspace: lane's 16 halves contiguous (32B).
__device__ inline v16h wn_load_b(const _Float16* __restrict__ wpk, int frag,
                                 int lane) {
  return *(const v16h*)(wpk + ((size_t)frag << 9) + (lane << 4));
}

// C/D fragment layout (32-bit 16x16): lane<16 -> N=lane, M=v (0..7);
// lane>=16 -> N=lane-16, M=v+8.
__device__ inline v8f wn_load_c(const float* __restrict__ src, int stride,
                                int col0, int lane) {
  const int n  = col0 + (lane & 15);
  const int mb = (lane >> 4) << 3;
  v8f c;
#pragma unroll
  for (int v = 0; v < 8; ++v) c[v] = src[(mb + v) * stride + n];
  return c;
}

__device__ inline void wn_store_d(float* __restrict__ dst, int stride, int col0,
                                  v8f d, float bias, int lane) {
  const int n  = col0 + (lane & 15);
  const int mb = (lane >> 4) << 3;
#pragma unroll
  for (int v = 0; v < 8; ++v) dst[(mb + v) * stride + n] = d[v] + bias;
}

// --------------------------- prep kernels ----------------------------------

__global__ void wn_zero_ring(float* __restrict__ ring, int n) {
  for (int i = blockIdx.x * blockDim.x + threadIdx.x; i < n;
       i += gridDim.x * blockDim.x)
    ring[i] = 0.0f;
}

// One block (32 lanes) per fragment: convert fp32 weights -> f16 packed in
// B-fragment register order (32x16 f16 B, column N=lane%16: lane<16 carries
// K=0..7,16..23, lane>=16 carries K=8..15,24..31, matching the A fragment).
__global__ void wn_pack(const float* __restrict__ cw,    // causal_w (L,2R,R,2)
                        const float* __restrict__ condw, // cond_w   (L,2R,C)
                        const float* __restrict__ rw,    // res_w    (L,R,R)
                        const float* __restrict__ sw,    // skip_w   (L,S,R)
                        const float* __restrict__ skcw,  // skipc_w  (S,R)
                        const float* __restrict__ fcw,   // fc_w     (S,S)
                        const float* __restrict__ cfw,   // condf_w  (S,C)
                        const float* __restrict__ logw,  // logits_w (Q,S)
                        _Float16* __restrict__ wpk) {
  const int f    = blockIdx.x;
  const int lane = threadIdx.x;
  const int n16  = lane & 15;
  const int half = lane >> 4;
  _Float16* dst = wpk + ((size_t)f << 9) + (lane << 4);

  int region, l = 0, tile = 0, kc = 0;
  int g = f;
  if (g < FR_RS)        { region = 0; l = g / 160; g %= 160; tile = g / 10; kc = g % 10; }
  else if (g < FR_SKC)  { region = 1; g -= FR_RS;  l = g / 64; g %= 64; tile = g / 4; kc = g % 4; }
  else if (g < FR_FC)   { region = 2; g -= FR_SKC; tile = g / 4; kc = g % 4; }
  else if (g < FR_LOG)  { region = 3; g -= FR_FC;  tile = g / 6; kc = g % 6; }
  else                  { region = 4; g -= FR_LOG; tile = g / 4; kc = g % 4; }

#pragma unroll
  for (int j = 0; j < 8; ++j) {
#pragma unroll
    for (int u = 0; u < 2; ++u) {
      const int kk = kc * 32 + half * 8 + ((j < 4) ? (2 * j + u) : (16 + 2 * (j - 4) + u));  // B-fragment K index
      float v = 0.0f;
      if (region == 0) {                 // z GEMM: K = [prev|res|cond] = 320
        const int n = tile * 16 + n16;   // 0..255
        if (kk < 128)       v = cw[(((size_t)l * 256 + n) * 128 + kk) * 2 + 0];
        else if (kk < 256)  v = cw[(((size_t)l * 256 + n) * 128 + (kk - 128)) * 2 + 1];
        else                v = condw[((size_t)l * 256 + n) * 64 + (kk - 256)];
      } else if (region == 1) {          // res (tiles 0-7) / skip (8-15)
        const int n = (tile & 7) * 16 + n16;
        v = (tile < 8) ? rw[((size_t)l * 128 + n) * 128 + kk]
                       : sw[((size_t)l * 128 + n) * 128 + kk];
      } else if (region == 2) {          // skip = res @ skipc_w.T
        v = skcw[(size_t)(tile * 16 + n16) * 128 + kk];
      } else if (region == 3) {          // s GEMM: K = [relu(skip)|cond] = 192
        const int n = tile * 16 + n16;
        v = (kk < 128) ? fcw[(size_t)n * 128 + kk]
                       : cfw[(size_t)n * 64 + (kk - 128)];
      } else {                           // logits
        v = logw[(size_t)(tile * 16 + n16) * 128 + kk];
      }
      dst[2 * j + u] = (_Float16)v;
    }
  }
}

// --------------------------- persistent generator --------------------------

__global__ __launch_bounds__(512) void wn_generate(
    const float* __restrict__ enc,      // (B,C,8)
    const float* __restrict__ first_w,  // (R,1,2)
    const float* __restrict__ first_b,  // (R)
    const float* __restrict__ causal_b, // (L,2R)
    const float* __restrict__ cond_b,   // (L,2R)
    const float* __restrict__ res_b,    // (L,R)
    const float* __restrict__ skip_b,   // (L,S)
    const float* __restrict__ skipc_b,  // (S)
    const float* __restrict__ fc_b,     // (S)
    const float* __restrict__ condf_b,  // (S)
    const float* __restrict__ logits_b, // (Q)
    const _Float16* __restrict__ wpk,   // packed fragments
    float* __restrict__ ring,           // dilation ring buffers
    int* __restrict__ out_samples,      // (T,B) int32
    float* __restrict__ out_logits) {   // (T,B,Q)
  __shared__ float pbuf[16 * RST];   // prev / h / head A-staging
  __shared__ float resb[16 * RST];   // residual state
  __shared__ float skipb[16 * RST];  // skip accumulator
  __shared__ float zbuf[16 * ZST];   // z / s / logits
  __shared__ float cbuf[16 * CST];   // cond (per step)
  __shared__ float bf[WN_B];         // buf_first
  __shared__ int   samp[WN_B];       // last sample (feedback)

  const int tid  = threadIdx.x;
  const int lane = tid & 31;
  const int wv   = tid >> 5;  // wave 0..15

  // LDS init: padding rows 8..15 must be zero and stay zero.
  for (int i = tid; i < 16 * RST; i += 512) {
    pbuf[i] = 0.0f; resb[i] = 0.0f; skipb[i] = 0.0f;
  }
  for (int i = tid; i < 16 * ZST; i += 512) zbuf[i] = 0.0f;
  for (int i = tid; i < 16 * CST; i += 512) cbuf[i] = 0.0f;
  if (tid < WN_B) { bf[tid] = 0.0f; samp[tid] = 128; }  // Q_ZERO
  __syncthreads();

  for (int t = 0; t < WN_T; ++t) {
    // ---- first layer (1x2 causal conv) + cond fetch -----------------------
    const int te = t >> 6;  // t / CR
    for (int i = tid; i < WN_B * WN_R; i += 512) {
      const int b = i >> 7, r = i & 127;
      const float x  = (float)samp[b] * (1.0f / 255.0f) - 0.5f;
      const float pv = bf[b];
      resb[b * RST + r] = pv * first_w[r * 2 + 0] + x * first_w[r * 2 + 1] + first_b[r];
    }
    for (int i = tid; i < WN_B * WN_C; i += 512) {
      const int b = i >> 6, cc = i & 63;
      cbuf[b * CST + cc] = enc[(b * WN_C + cc) * 8 + te];
    }
    // warm layer-0 ring line while we compute
    if (tid < 32) __builtin_prefetch(ring + tid * 32, 0, 0);
    __syncthreads();
    if (tid < WN_B) bf[tid] = (float)samp[tid] * (1.0f / 255.0f) - 0.5f;

    // ---- skip = res @ skipc_w.T + skipc_b (waves 0..7) --------------------
    if (wv < 8) {
      v8f acc = {};
#pragma unroll
      for (int kc = 0; kc < 4; ++kc)
        acc = wn_wmma(wn_build_a(resb, RST, kc * 32, lane),
                      wn_load_b(wpk, FR_SKC + wv * 4 + kc, lane), acc);
      const float bias = (lane < 16) ? skipc_b[wv * 16 + (lane & 15)] : 0.0f;
      wn_store_d(skipb, RST, wv * 16, acc, bias, lane);
    }
    __syncthreads();

    // ---- 20 dilated residual layers ---------------------------------------
    for (int l = 0; l < WN_L; ++l) {
      const int d   = kDIL[l];
      const int idx = t & (d - 1);
      float* ringp = ring + (size_t)kROFF[l] * (WN_B * WN_R) + (size_t)idx * (WN_B * WN_R);

      // prev <- ring ; ring <- res (incoming residual)
      for (int i = tid; i < WN_B * WN_R; i += 512) {
        const int b = i >> 7, r = i & 127;
        const float pv = ringp[i];
        ringp[i] = resb[b * RST + r];
        pbuf[b * RST + r] = pv;
      }
      // warm this layer's first z fragment (one line per lane, per wave)
      __builtin_prefetch(wpk + ((size_t)((l * 16 + wv) * 10) << 9) + (lane << 4), 0, 0);
      __syncthreads();

      // z = [prev|res|cond] @ W.T + (causal_b + cond_b)  -- K=320, N=256
      {
        // warm this layer's res/skip fragments for the phase after next
        __builtin_prefetch(
            wpk + ((size_t)(FR_RS + (l * 16 + wv) * 4) << 9) + (lane << 4), 0, 0);
        v8f acc = {};
#pragma unroll
        for (int kc = 0; kc < 10; ++kc) {
          v16h a = (kc < 4)   ? wn_build_a(pbuf, RST, kc * 32, lane)
                 : (kc < 8)   ? wn_build_a(resb, RST, (kc - 4) * 32, lane)
                              : wn_build_a(cbuf, CST, (kc - 8) * 32, lane);
          acc = wn_wmma(a, wn_load_b(wpk, (l * 16 + wv) * 10 + kc, lane), acc);
        }
        const int n = wv * 16 + (lane & 15);
        const float bias =
            (lane < 16) ? (causal_b[l * 256 + n] + cond_b[l * 256 + n]) : 0.0f;
        wn_store_d(zbuf, ZST, wv * 16, acc, bias, lane);
      }
      __syncthreads();

      // h = sigmoid(gate) * tanh(out) -> pbuf[:,0:128]
      for (int i = tid; i < WN_B * WN_R; i += 512) {
        const int b = i >> 7, j = i & 127;
        const float g = zbuf[b * ZST + j];
        const float o = zbuf[b * ZST + 128 + j];
        pbuf[b * RST + j] = (1.0f / (1.0f + __expf(-g))) * tanhf(o);
      }
      // warm next layer's ring line (used two phases from now)
      if (l + 1 < WN_L && tid < 32) {
        const int dn   = kDIL[l + 1];
        const int idxn = t & (dn - 1);
        const float* nring = ring + (size_t)kROFF[l + 1] * (WN_B * WN_R) +
                             (size_t)idxn * (WN_B * WN_R);
        __builtin_prefetch(nring + tid * 32, 0, 0);
      }
      __syncthreads();

      // res += h @ res_w.T + res_b (waves 0-7); skip += h @ skip_w.T (8-15)
      {
        float* dstb = (wv < 8) ? resb : skipb;
        const int tile = wv & 7;
        const float* bp = (wv < 8) ? (res_b + l * 128) : (skip_b + l * 128);
        v8f acc = wn_load_c(dstb, RST, tile * 16, lane);
#pragma unroll
        for (int kc = 0; kc < 4; ++kc)
          acc = wn_wmma(wn_build_a(pbuf, RST, kc * 32, lane),
                        wn_load_b(wpk, FR_RS + (l * 16 + wv) * 4 + kc, lane), acc);
        const float bias = (lane < 16) ? bp[tile * 16 + (lane & 15)] : 0.0f;
        wn_store_d(dstb, RST, tile * 16, acc, bias, lane);
      }
      __syncthreads();
    }

    // ---- head: s = relu(skip)@fc.T + c@condf.T + biases -------------------
    for (int i = tid; i < WN_B * WN_S; i += 512) {
      const int b = i >> 7, j = i & 127;
      pbuf[b * RST + j] = fmaxf(skipb[b * RST + j], 0.0f);
    }
    if (wv < 8)
      __builtin_prefetch(wpk + ((size_t)(FR_FC + wv * 6) << 9) + (lane << 4), 0, 0);
    __syncthreads();
    if (wv < 8) {  // K = [relu(skip)(128) | cond(64)], N = 128
      v8f acc = {};
#pragma unroll
      for (int kc = 0; kc < 6; ++kc) {
        v16h a = (kc < 4) ? wn_build_a(pbuf, RST, kc * 32, lane)
                          : wn_build_a(cbuf, CST, (kc - 4) * 32, lane);
        acc = wn_wmma(a, wn_load_b(wpk, FR_FC + wv * 6 + kc, lane), acc);
      }
      const int n = wv * 16 + (lane & 15);
      const float bias = (lane < 16) ? (fc_b[n] + condf_b[n]) : 0.0f;
      wn_store_d(zbuf, ZST, wv * 16, acc, bias, lane);
    }
    __builtin_prefetch(wpk + ((size_t)(FR_LOG + wv * 4) << 9) + (lane << 4), 0, 0);
    __syncthreads();
    for (int i = tid; i < WN_B * WN_S; i += 512) {  // relu(s)
      const int b = i >> 7, j = i & 127;
      pbuf[b * RST + j] = fmaxf(zbuf[b * ZST + j], 0.0f);
    }
    __syncthreads();
    {  // logits = relu(s) @ logits_w.T + logits_b  -- K=128, N=256, 16 waves
      v8f acc = {};
#pragma unroll
      for (int kc = 0; kc < 4; ++kc)
        acc = wn_wmma(wn_build_a(pbuf, RST, kc * 32, lane),
                      wn_load_b(wpk, FR_LOG + wv * 4 + kc, lane), acc);
      const int n = wv * 16 + (lane & 15);
      const float bias = (lane < 16) ? logits_b[n] : 0.0f;
      wn_store_d(zbuf, ZST, wv * 16, acc, bias, lane);
    }
    __syncthreads();

    // ---- emit logits, wave32 argmax (first-index tie-break), feedback -----
    for (int i = tid; i < WN_B * WN_Q; i += 512) {
      const int b = i >> 8, q = i & 255;
      out_logits[((size_t)t * WN_B + b) * WN_Q + q] = zbuf[b * ZST + q];
    }
    if (wv < 8) {
      const int b = wv;
      float bv = -3.402823466e38f;
      int   bi = 0;
      for (int q = lane; q < WN_Q; q += 32) {
        const float v = zbuf[b * ZST + q];
        if (v > bv) { bv = v; bi = q; }   // ascending q keeps first max
      }
#pragma unroll
      for (int off = 16; off >= 1; off >>= 1) {
        const float ov = __shfl_down(bv, off, 32);
        const int   oi = __shfl_down(bi, off, 32);
        if (ov > bv || (ov == bv && oi < bi)) { bv = ov; bi = oi; }
      }
      if (lane == 0) {
        samp[b] = bi;
        out_samples[t * WN_B + b] = bi;
      }
    }
    __syncthreads();
  }
}

// --------------------------- launcher --------------------------------------

extern "C" void kernel_launch(void* const* d_in, const int* in_sizes, int n_in,
                              void* d_out, int out_size, void* d_ws,
                              size_t ws_size, hipStream_t stream) {
  const float* enc      = (const float*)d_in[0];
  const float* first_w  = (const float*)d_in[1];
  const float* first_b  = (const float*)d_in[2];
  const float* causal_w = (const float*)d_in[3];
  const float* causal_b = (const float*)d_in[4];
  const float* cond_w   = (const float*)d_in[5];
  const float* cond_b   = (const float*)d_in[6];
  const float* res_w    = (const float*)d_in[7];
  const float* res_b    = (const float*)d_in[8];
  const float* skip_w   = (const float*)d_in[9];
  const float* skip_b   = (const float*)d_in[10];
  const float* skipc_w  = (const float*)d_in[11];
  const float* skipc_b  = (const float*)d_in[12];
  const float* fc_w     = (const float*)d_in[13];
  const float* fc_b     = (const float*)d_in[14];
  const float* condf_w  = (const float*)d_in[15];
  const float* condf_b  = (const float*)d_in[16];
  const float* logits_w = (const float*)d_in[17];
  const float* logits_b = (const float*)d_in[18];

  float*     ring = (float*)d_ws;
  _Float16*  wpk  = (_Float16*)((char*)d_ws + RING_BYTES);

  int*   out_samples = (int*)d_out;                    // (T,B) int32
  float* out_logits  = (float*)d_out + WN_T * WN_B;    // (T,B,Q) f32

  wn_zero_ring<<<2048, 256, 0, stream>>>(ring, RING_FLOATS);
  wn_pack<<<NFRAG, 32, 0, stream>>>(causal_w, cond_w, res_w, skip_w, skipc_w,
                                    fc_w, condf_w, logits_w, wpk);
  wn_generate<<<1, 512, 0, stream>>>(enc, first_w, first_b, causal_b, cond_b,
                                     res_b, skip_b, skipc_b, fc_b, condf_b,
                                     logits_b, wpk, ring, out_samples,
                                     out_logits);
}
